// AdaRNN_77524159693110
// MI455X (gfx1250) — matrix operand order repaired and checked by reading, not compiled, not run on hardware
//
#include <hip/hip_runtime.h>
#include <cstdint>

// ---------- types / helpers ----------
typedef __bf16 bf16_t;
typedef bf16_t v16bf __attribute__((ext_vector_type(16)));
typedef bf16_t v8bf  __attribute__((ext_vector_type(8)));
typedef float  v8f   __attribute__((ext_vector_type(8)));

struct alignas(16) U4 { uint32_t x, y, z, w; };
struct alignas(32) U8 { U4 a, b; };
struct alignas(16) F4 { float x, y, z, w; };

__device__ __forceinline__ v8bf cvt8v(F4 p, F4 q) {   // native f32->bf16 converts
  v8bf r;
  r[0] = (bf16_t)p.x; r[1] = (bf16_t)p.y; r[2] = (bf16_t)p.z; r[3] = (bf16_t)p.w;
  r[4] = (bf16_t)q.x; r[5] = (bf16_t)q.y; r[6] = (bf16_t)q.z; r[7] = (bf16_t)q.w;
  return r;
}
// fragment from bf16 memory: 8 bf16 at k0 and 8 bf16 at k1 (each 16B aligned)
__device__ __forceinline__ v16bf frag_bf(const bf16_t* base, int k0, int k1) {
  U8 u; u.a = *(const U4*)(base + k0); u.b = *(const U4*)(base + k1);
  return __builtin_bit_cast(v16bf, u);
}
__device__ __forceinline__ v8f wmma_bf16(v16bf a, v16bf b, v8f c) {
  return __builtin_amdgcn_wmma_f32_16x16x32_bf16(false, a, false, b, (short)0, c,
                                                 false, false);
}
__device__ __forceinline__ v8f zero8() {
  v8f z = {0.f,0.f,0.f,0.f,0.f,0.f,0.f,0.f}; return z;
}
__device__ __forceinline__ float sigmoidf_(float x) { return 1.0f / (1.0f + expf(-x)); }

// ---------- vectorized f32 -> bf16 convert ----------
__global__ __launch_bounds__(256) void cvt_bf16_kernel(
    const float* __restrict__ in, bf16_t* __restrict__ out, int n8) {
  int i = blockIdx.x * 256 + threadIdx.x;
  if (i < n8) {
    long b = (long)i * 8;
    F4 p = *(const F4*)(in + b), q = *(const F4*)(in + b + 4);
    *(U4*)(out + b) = __builtin_bit_cast(U4, cvt8v(p, q));
  }
}

// ---------- GEMM: C[M,N] = A[M,K]bf16 @ B[N,K]^T bf16 + bias, f32 accum --------
// NB = N-tiles per wave (shared A fragment). Tail columns clamp B row to 0 and
// are simply never stored (no in-loop predication).
template <int NB>
__global__ __launch_bounds__(256) void gemm_bias_bf16_kernel(
    const bf16_t* __restrict__ A, long strideA, long offA,
    const bf16_t* __restrict__ B, const float* __restrict__ bias,
    float* __restrict__ C, int Mtiles, int NtileGroups, int N, int K, int ldc) {
  int gw = blockIdx.x * 8 + (threadIdx.x >> 5);
  if (gw >= Mtiles * NtileGroups) return;
  int tm = gw / NtileGroups, tg = gw % NtileGroups;
  int lane = threadIdx.x & 31;
  const bf16_t* aRow = A + offA + (long)(tm * 16 + (lane & 15)) * strideA;
  int ncol[NB];
  const bf16_t* bRow[NB];
  v8f acc[NB];
#pragma unroll
  for (int c = 0; c < NB; ++c) {
    ncol[c] = (tg * NB + c) * 16 + (lane & 15);
    bRow[c] = B + (long)(ncol[c] < N ? ncol[c] : 0) * K;   // clamp, discard later
    acc[c] = zero8();
  }
  for (int kc = 0; kc < K; kc += 32) {
    int ka = kc + ((lane >> 4) << 3);
    int kb = kc + ((lane >> 4) << 3);
    v16bf af = frag_bf(aRow, ka, ka + 16);
#pragma unroll
    for (int c = 0; c < NB; ++c)
      acc[c] = wmma_bf16(af, frag_bf(bRow[c], kb, kb + 16), acc[c]);
  }
#pragma unroll
  for (int c = 0; c < NB; ++c) {
    if (ncol[c] < N) {
      float bv = bias[ncol[c]];
#pragma unroll
      for (int r = 0; r < 8; ++r) {
        int gm = tm * 16 + ((lane >> 4) << 3) + r;
        C[(long)gm * ldc + ncol[c]] = acc[c][r] + bv;
      }
    }
  }
}

// ---------- GRU scan: 32 WGs x 16 batch rows, 8 waves (one 16x16 h-tile each) ----
// xw: (B,S,3H) f32 = x@Wih^T+bih ; whh: (3H,H) f32 (staged to LDS bf16) ;
// obf: (B,S,H) bf16 output. Dynamic LDS: whh 384*128 bf16 + h 16*128 bf16.
__global__ __launch_bounds__(256) void gru_scan_kernel(
    const float* __restrict__ xw, const float* __restrict__ whh,
    const float* __restrict__ bhh, bf16_t* __restrict__ obf) {
  extern __shared__ char smem[];
  bf16_t* wl  = (bf16_t*)smem;            // 384*128 bf16  = 96KB
  bf16_t* hsh = wl + 384 * 128;           // 16*128  bf16  = 4KB
  const int lane = threadIdx.x & 31;
  const int wave = threadIdx.x >> 5;
  const int b0 = blockIdx.x * 16;
  // stage whh -> LDS as bf16 (24 x 8 elements per thread)
  for (int idx = threadIdx.x * 8; idx < 384 * 128; idx += 256 * 8) {
    F4 p = *(const F4*)(whh + idx), q = *(const F4*)(whh + idx + 4);
    *(U4*)(wl + idx) = __builtin_bit_cast(U4, cvt8v(p, q));
  }
  { U4 z = {}; *(U4*)(hsh + threadIdx.x * 8) = z; }      // zero hidden state
  const int ncol = wave * 16 + (lane & 15);              // h feature
  const float br = bhh[ncol], bz = bhh[128 + ncol], bn = bhh[256 + ncol];
  const int ilocal = ((lane >> 4) << 3);
  float hreg[8];
#pragma unroll
  for (int r = 0; r < 8; ++r) hreg[r] = 0.0f;
  const bf16_t* bR = wl + (long)ncol * 128;
  const bf16_t* bZ = wl + (long)(128 + ncol) * 128;
  const bf16_t* bN = wl + (long)(256 + ncol) * 128;
  const bf16_t* aRow = hsh + (lane & 15) * 128;
  __syncthreads();
#pragma unroll 1
  for (int s = 0; s < 128; ++s) {
    v8f accR = zero8(), accZ = zero8(), accN = zero8();
#pragma unroll
    for (int kc = 0; kc < 128; kc += 32) {
      int ka = kc + ((lane >> 4) << 3);
      v16bf a = frag_bf(aRow, ka, ka + 16);              // LDS
      int kb = kc + ((lane >> 4) << 3);
      accR = wmma_bf16(a, frag_bf(bR, kb, kb + 16), accR);
      accZ = wmma_bf16(a, frag_bf(bZ, kb, kb + 16), accZ);
      accN = wmma_bf16(a, frag_bf(bN, kb, kb + 16), accN);
    }
    __syncthreads();                                     // hsh reads done
#pragma unroll
    for (int r = 0; r < 8; ++r) {
      int i = ilocal + r;
      long base = ((long)(b0 + i) * 128 + s) * 384;
      if (s + 1 < 128)                                   // global_prefetch_b8
        __builtin_prefetch(xw + base + 384 + ncol, 0, 0);
      float xr = xw[base + ncol], xz = xw[base + 128 + ncol], xn = xw[base + 256 + ncol];
      float rg = sigmoidf_(xr + accR[r] + br);
      float zg = sigmoidf_(xz + accZ[r] + bz);
      float ng = tanhf(xn + rg * (accN[r] + bn));
      float hnew = (1.0f - zg) * ng + zg * hreg[r];
      hreg[r] = hnew;
      bf16_t hb = (bf16_t)hnew;
      obf[((long)(b0 + i) * 128 + s) * 128 + ncol] = hb;
      hsh[i * 128 + ncol] = hb;
    }
    __syncthreads();                                     // hsh writes visible
  }
}

// ---------- gate GEMM: y[256,128] = gather(o_bf)[256,32768] @ gw_bf^T + gb ------
// 4 N-tiles per wave sharing the gathered A fragment.
__device__ __forceinline__ U4 gatherA_bf(const bf16_t* __restrict__ o, int i, int k) {
  int s = k >> 8, j = k & 255;
  int b = (j < 128) ? i : (i + 256);
  return *(const U4*)(o + ((long)b * 128 + s) * 128 + (j & 127));
}
__global__ __launch_bounds__(256) void gate_gemm_kernel(
    const bf16_t* __restrict__ o, const bf16_t* __restrict__ gw,
    const float* __restrict__ gb, float* __restrict__ y) {
  int gwid = blockIdx.x * 8 + (threadIdx.x >> 5);        // 0..31
  int tm = gwid >> 1, tg = gwid & 1;                     // 16 M-tiles x 2 groups
  int lane = threadIdx.x & 31;
  int i = tm * 16 + (lane & 15);
  int ncol[4];
  const bf16_t* bRow[4];
  v8f acc[4];
#pragma unroll
  for (int c = 0; c < 4; ++c) {
    ncol[c] = (tg * 4 + c) * 16 + (lane & 15);
    bRow[c] = gw + (long)ncol[c] * 32768;
    acc[c] = zero8();
  }
#pragma unroll 1
  for (int kc = 0; kc < 32768; kc += 32) {
    int k0 = kc + ((lane >> 4) << 3);
    U8 ua; ua.a = gatherA_bf(o, i, k0); ua.b = gatherA_bf(o, i, k0 + 16);
    v16bf af = __builtin_bit_cast(v16bf, ua);
    int kb = kc + ((lane >> 4) << 3);
#pragma unroll
    for (int c = 0; c < 4; ++c)
      acc[c] = wmma_bf16(af, frag_bf(bRow[c], kb, kb + 16), acc[c]);
  }
#pragma unroll
  for (int c = 0; c < 4; ++c) {
    float bv = gb[ncol[c]];
#pragma unroll
    for (int r = 0; r < 8; ++r) {
      int gm = tm * 16 + ((lane >> 4) << 3) + r;
      y[(long)gm * 128 + ncol[c]] = acc[c][r] + bv;
    }
  }
}

// ---------- gate finalize: BN(axis 0) -> sigmoid -> mean -> softmax ----------
__global__ __launch_bounds__(128) void gate_finalize_kernel(
    const float* __restrict__ y, const float* __restrict__ gamma,
    const float* __restrict__ beta, float* __restrict__ wout) {
  __shared__ float sh[128];
  int n = threadIdx.x;
  float sum = 0.f, sumsq = 0.f;
  for (int i = 0; i < 256; ++i) { float v = y[i * 128 + n]; sum += v; sumsq += v * v; }
  float m = sum * (1.0f / 256.0f);
  float var = sumsq * (1.0f / 256.0f) - m * m;
  float sc = gamma[n] * rsqrtf(var + 1e-5f);
  float bt = beta[n];
  float sacc = 0.f;
  for (int i = 0; i < 256; ++i)
    sacc += sigmoidf_(sc * (y[i * 128 + n] - m) + bt);
  float w = sacc * (1.0f / 256.0f);
  sh[n] = w; __syncthreads();
  for (int st = 64; st > 0; st >>= 1) { if (n < st) sh[n] = fmaxf(sh[n], sh[n + st]); __syncthreads(); }
  float mx = sh[0]; __syncthreads();
  float e = expf(w - mx);
  sh[n] = e; __syncthreads();
  for (int st = 64; st > 0; st >>= 1) { if (n < st) sh[n] += sh[n + st]; __syncthreads(); }
  wout[n] = e / sh[0];
}

// ---------- MMD: one WG per (layer, s); Gram via WMMA from LDS bf16 ----------
__global__ __launch_bounds__(256) void mmd_kernel(
    const bf16_t* __restrict__ o1, const bf16_t* __restrict__ o2,
    float* __restrict__ mmd_out) {
  extern __shared__ char smem[];
  bf16_t* t  = (bf16_t*)smem;                       // 512*128 bf16 = 128KB
  float*  sq = (float*)(smem + 512 * 128 * 2);      // 512
  float* red = sq + 512;                            // 256
  float* scal = red + 256;                          // 1
  int l = blockIdx.x >> 7, s = blockIdx.x & 127;
  const bf16_t* o = l ? o2 : o1;
  int tid = threadIdx.x, lane = tid & 31, wave = tid >> 5;
  // stage rows (bf16 vector copy) + row sq-norms
#pragma unroll
  for (int rr = 0; rr < 2; ++rr) {
    int b = tid * 2 + rr;
    const bf16_t* p = o + ((long)b * 128 + s) * 128;
    float a = 0.f;
    for (int k = 0; k < 128; k += 8) {
      U4 d = *(const U4*)(p + k);
      *(U4*)(t + b * 128 + k) = d;
      v8bf v = __builtin_bit_cast(v8bf, d);
#pragma unroll
      for (int e = 0; e < 8; ++e) { float f = (float)v[e]; a += f * f; }
    }
    sq[b] = a;
  }
  __syncthreads();
  if (tid < 128) {                                  // column sums -> sumG = |colsum|^2
    float cs = 0.f;
    for (int b = 0; b < 512; ++b) cs += (float)o[((long)b * 128 + s) * 128 + tid];
    red[tid] = cs * cs;
  }
  __syncthreads();
  if (tid == 0) {
    float sg = 0.f; for (int k = 0; k < 128; ++k) sg += red[k];
    float ssq = 0.f; for (int b = 0; b < 512; ++b) ssq += sq[b];
    float d2sum = 2.0f * 512.0f * ssq - 2.0f * sg;
    float bw = d2sum / (512.0f * 511.0f) / 4.0f;    // / KM^(KN//2)
    scal[0] = 1.0f / bw;
  }
  __syncthreads();
  float invbw = scal[0];
  float local = 0.0f;
  for (int tile = wave; tile < 1024; tile += 8) {
    int ti = tile >> 5, tj = tile & 31;
    const bf16_t* aRow = t + (ti * 16 + (lane & 15)) * 128;
    const bf16_t* bRow = t + (tj * 16 + (lane & 15)) * 128;
    v8f acc = zero8();
#pragma unroll
    for (int kc = 0; kc < 128; kc += 32) {
      int ka = kc + ((lane >> 4) << 3);
      int kb = kc + ((lane >> 4) << 3);
      acc = wmma_bf16(frag_bf(aRow, ka, ka + 16), frag_bf(bRow, kb, kb + 16), acc);
    }
    int j = tj * 16 + (lane & 15);
    float sqj = sq[j];
    bool jy = (j >= 256);
#pragma unroll
    for (int r = 0; r < 8; ++r) {
      int i = ti * 16 + ((lane >> 4) << 3) + r;
      float d2 = sq[i] + sqj - 2.0f * acc[r];
      float e = invbw * d2;
      float ks = expf(-e) + expf(-0.5f * e) + expf(-0.25f * e)
               + expf(-0.125f * e) + expf(-0.0625f * e);
      local += (((i >= 256) == jy) ? ks : -ks);
    }
  }
  red[tid] = local; __syncthreads();
  for (int st = 128; st > 0; st >>= 1) { if (tid < st) red[tid] += red[tid + st]; __syncthreads(); }
  if (tid == 0) mmd_out[blockIdx.x] = red[0] * (1.0f / 65536.0f);
}

// ---------- final loss: sum(weights * mmd) ----------
__global__ __launch_bounds__(256) void loss_kernel(
    const float* __restrict__ mmd, const float* __restrict__ weights,
    float* __restrict__ out) {
  __shared__ float sh[256];
  int t = threadIdx.x;
  sh[t] = mmd[t] * weights[t];
  __syncthreads();
  for (int st = 128; st > 0; st >>= 1) { if (t < st) sh[t] += sh[t + st]; __syncthreads(); }
  if (t == 0) out[0] = sh[0];
}

// ---------- host launch ----------
extern "C" void kernel_launch(void* const* d_in, const int* in_sizes, int n_in,
                              void* d_out, int out_size, void* d_ws, size_t ws_size,
                              hipStream_t stream) {
  const float* x    = (const float*)d_in[0];
  const float* wih0 = (const float*)d_in[1];
  const float* whh0 = (const float*)d_in[2];
  const float* bih0 = (const float*)d_in[3];
  const float* bhh0 = (const float*)d_in[4];
  const float* wih1 = (const float*)d_in[5];
  const float* whh1 = (const float*)d_in[6];
  const float* bih1 = (const float*)d_in[7];
  const float* bhh1 = (const float*)d_in[8];
  const float* gw0  = (const float*)d_in[9];
  const float* gb0  = (const float*)d_in[10];
  const float* bg0  = (const float*)d_in[11];
  const float* bb0  = (const float*)d_in[12];
  const float* gw1  = (const float*)d_in[13];
  const float* gb1  = (const float*)d_in[14];
  const float* bg1  = (const float*)d_in[15];
  const float* bb1  = (const float*)d_in[16];
  const float* fcw  = (const float*)d_in[17];
  const float* fcb  = (const float*)d_in[18];
  (void)in_sizes; (void)n_in; (void)out_size; (void)ws_size;

  float* out = (float*)d_out;                    // [fc 3072][loss 1][weights 256]
  // f32 region
  float* ws  = (float*)d_ws;
  float* xw  = ws;                               // 512*128*384
  float* yb  = xw + (size_t)512 * 128 * 384;     // 256*128
  float* mmdbuf = yb + 256 * 128;                // 256
  // bf16 region
  bf16_t* xbf   = (bf16_t*)(mmdbuf + 256);
  bf16_t* o1bf  = xbf  + (size_t)512 * 128 * 128;
  bf16_t* o2bf  = o1bf + (size_t)512 * 128 * 128;
  bf16_t* wih0b = o2bf + (size_t)512 * 128 * 128;
  bf16_t* wih1b = wih0b + 384 * 128;
  bf16_t* fcwb  = wih1b + 384 * 128;
  bf16_t* gwb   = fcwb + 768;                    // 128*32768, reused per layer

  size_t scanShm = (size_t)(384 * 128 + 16 * 128) * sizeof(bf16_t);   // 100KB
  size_t mmdShm  = (size_t)512 * 128 * 2 + 512 * 4 + 256 * 4 + 16;    // ~131KB
  hipFuncSetAttribute((const void*)gru_scan_kernel,
                      hipFuncAttributeMaxDynamicSharedMemorySize, (int)scanShm);
  hipFuncSetAttribute((const void*)mmd_kernel,
                      hipFuncAttributeMaxDynamicSharedMemorySize, (int)mmdShm);

  // one-time f32 -> bf16 staging
  cvt_bf16_kernel<<<4096, 256, 0, stream>>>(x, xbf, 1048576);        // 512*128*128/8
  cvt_bf16_kernel<<<24, 256, 0, stream>>>(wih0, wih0b, 6144);
  cvt_bf16_kernel<<<24, 256, 0, stream>>>(wih1, wih1b, 6144);
  cvt_bf16_kernel<<<1, 256, 0, stream>>>(fcw, fcwb, 96);

  // layer 0: xw = x @ wih0^T + bih0  (4096 M-tiles x 6 groups of 4 N-tiles)
  gemm_bias_bf16_kernel<4><<<3072, 256, 0, stream>>>(xbf, 128L, 0L, wih0b, bih0, xw,
                                                     4096, 6, 384, 128, 384);
  gru_scan_kernel<<<32, 256, scanShm, stream>>>(xw, whh0, bhh0, o1bf);
  // layer 1
  gemm_bias_bf16_kernel<4><<<3072, 256, 0, stream>>>(o1bf, 128L, 0L, wih1b, bih1, xw,
                                                     4096, 6, 384, 128, 384);
  gru_scan_kernel<<<32, 256, scanShm, stream>>>(xw, whh1, bhh1, o2bf);
  // gate weights
  cvt_bf16_kernel<<<2048, 256, 0, stream>>>(gw0, gwb, 524288);
  gate_gemm_kernel<<<4, 256, 0, stream>>>(o1bf, gwb, gb0, yb);
  gate_finalize_kernel<<<1, 128, 0, stream>>>(yb, bg0, bb0, out + 3073);
  cvt_bf16_kernel<<<2048, 256, 0, stream>>>(gw1, gwb, 524288);
  gate_gemm_kernel<<<4, 256, 0, stream>>>(o2bf, gwb, gb1, yb);
  gate_finalize_kernel<<<1, 128, 0, stream>>>(yb, bg1, bb1, out + 3073 + 128);
  // fc head: A rows = o2[b,127,:], N=6 tail via clamped B row + guarded store
  gemm_bias_bf16_kernel<1><<<4, 256, 0, stream>>>(o2bf, 16384L, (long)(127 * 128),
                                                  fcwb, fcb, out, 32, 1, 6, 128, 6);
  // MMD per (layer, s)
  mmd_kernel<<<256, 256, mmdShm, stream>>>(o1bf, o2bf, mmdbuf);
  // loss_transfer
  loss_kernel<<<1, 256, 0, stream>>>(mmdbuf, out + 3073, out + 3072);
}
